// Qwen3Attention_63170378989896
// MI455X (gfx1250) — matrix operand order repaired and checked by reading, not compiled, not run on hardware
//
#include <hip/hip_runtime.h>
#include <hip/hip_bf16.h>

// ---------------------------------------------------------------------------
// Qwen3 attention block for gfx1250 (MI455X), wave32 + WMMA f16 (fp32 accum).
//   B=2 T=2048 HIDDEN=2048 NH=16 NKV=8 HD=128, causal GQA, QK-RMSNorm, RoPE.
// Pipeline: f32->f16 convert -> QKV GEMM (WMMA, LDS-staged A w/ async copy)
//           -> norm+rope -> flash attn (WMMA QK^T / PV, online softmax)
//           -> O-proj GEMM.
// ---------------------------------------------------------------------------

#define BB    2
#define TT    2048
#define HID   2048
#define NH    16
#define NKV   8
#define GG    2
#define HD    128
#define QS    (NH * HD)       // 2048
#define KVS   (NKV * HD)      // 1024
#define QKVO  (QS + 2 * KVS)  // 4096
#define MROWS (BB * TT)       // 4096

typedef __attribute__((ext_vector_type(8)))  _Float16 v8h;
typedef __attribute__((ext_vector_type(16))) _Float16 v16h;
typedef __attribute__((ext_vector_type(8)))  float    v8f;
typedef __attribute__((ext_vector_type(4)))  int      v4i;

// ---- gfx1250 async global->LDS copy (guarded; falls back to sync bounce) ----
#if defined(__has_builtin)
#  if __has_builtin(__builtin_amdgcn_global_load_async_to_lds_b128) && \
      __has_builtin(__builtin_amdgcn_s_wait_asynccnt)
#    define USE_ASYNC_LDS 1
#  endif
#endif
#ifndef USE_ASYNC_LDS
#  define USE_ASYNC_LDS 0
#endif

__device__ inline void async_wait_all() {
#if USE_ASYNC_LDS
    __builtin_amdgcn_s_wait_asynccnt(0);
#endif
}

__device__ inline void copy16_g2l(const _Float16* g, _Float16* l) {
#if USE_ASYNC_LDS
    __builtin_amdgcn_global_load_async_to_lds_b128(
        (__attribute__((address_space(1))) v4i*)g,
        (__attribute__((address_space(3))) v4i*)l, 0, 0);
#else
    *(v8h*)l = *(const v8h*)g;
#endif
}

__device__ inline v8f vzero8f() {
    v8f z = {0.f, 0.f, 0.f, 0.f, 0.f, 0.f, 0.f, 0.f};
    return z;
}

// D = A*B + C, A 16x32 f16, B 32x16 f16, C/D 16x16 f32 (wave32 WMMA)
__device__ inline v8f wmma16x16x32(v16h a, v16h b, v8f c) {
    return __builtin_amdgcn_wmma_f32_16x16x32_f16(
        /*neg_a=*/false, a, /*neg_b=*/false, b,
        /*c_mod=*/(short)0, c, /*reuse_a=*/false, /*reuse_b=*/false);
}

// A-fragment (16x32, M x K) from row-major [*, ld] f16, tile at (row0, k0).
// ISA layout: lane -> M = lane&15; lanes 0-15 K-base 0, lanes 16-31 K-base 8;
// elements 0..7 -> K = kb+0..7, elements 8..15 -> K = 16+kb+0..7.
__device__ inline v16h load_a_frag(const _Float16* base, int ld, int row0,
                                   int k0, int lane) {
    int m  = lane & 15;
    int kb = (lane >> 4) << 3;  // 0 or 8
    const _Float16* p = base + (size_t)(row0 + m) * ld + k0 + kb;
    v8h lo = *(const v8h*)(p);
    v8h hi = *(const v8h*)(p + 16);
    v16h f;
#pragma unroll
    for (int i = 0; i < 8; ++i) { f[i] = lo[i]; f[i + 8] = hi[i]; }
    return f;
}

// Same A-fragment mapping, from an LDS row-major tile with stride 32 f16.
__device__ inline v16h load_a_frag_lds(const _Float16* base, int row0,
                                       int lane) {
    int m  = lane & 15;
    int kb = (lane >> 4) << 3;
    const _Float16* p = base + (row0 + m) * 32 + kb;
    v8h lo = *(const v8h*)(p);
    v8h hi = *(const v8h*)(p + 16);
    v16h f;
#pragma unroll
    for (int i = 0; i < 8; ++i) { f[i] = lo[i]; f[i + 8] = hi[i]; }
    return f;
}

// B-fragment (32x16, K x N) where B(k,n) = Bm[n0+n][k0+k], Bm row-major [*, ld].
// ISA layout: lane -> N = lane&15; K-base kb = 8*(lane>=16);
// elements 0..7 -> K = kb+0..7, elements 8..15 -> K = 16+kb+0..7.
__device__ inline v16h load_b_frag(const _Float16* base, int ld, int n0,
                                   int k0, int lane) {
    const _Float16* p =
        base + (size_t)(n0 + (lane & 15)) * ld + k0 + ((lane >> 4) << 3);
    v8h lo = *(const v8h*)(p);
    v8h hi = *(const v8h*)(p + 16);
    v16h f;
#pragma unroll
    for (int i = 0; i < 8; ++i) { f[i] = lo[i]; f[i + 8] = hi[i]; }
    return f;
}

// ---------------------------------------------------------------------------
// Kernel 1: f32 -> f16 conversion (grid-stride)
// ---------------------------------------------------------------------------
__global__ void f32_to_f16_kernel(const float* __restrict__ src,
                                  _Float16* __restrict__ dst, int n) {
    for (int i = blockIdx.x * blockDim.x + threadIdx.x; i < n;
         i += gridDim.x * blockDim.x)
        dst[i] = (_Float16)src[i];
}

// ---------------------------------------------------------------------------
// Kernel 2/5: C[M,N] (f32) = A[M,K] (f16, row-major) * Bm[N,K]^T (f16)
// Block = 8 waves -> 128x128 block tile; wave tile 64x32 (4x2 fragments,
// 8 WMMAs per K-step of 32). A tile (128x32 = 8KB) is double-buffered in LDS
// and streamed with async global->LDS copies; B read direct (L2-resident).
// Requires M%128==0, N%128==0, K%32==0.
// ---------------------------------------------------------------------------
__global__ __launch_bounds__(256) void gemm_nt_f16_kernel(
    const _Float16* __restrict__ A, const _Float16* __restrict__ Bm,
    float* __restrict__ C, int K, int lda, int ldb, int ldc) {
    __shared__ _Float16 lbufA[2][128 * 32];  // 2 x 8KB

    int tid  = threadIdx.x;
    int lane = tid & 31;
    int wave = tid >> 5;
    int rowbase = blockIdx.y * 128;
    int r0 = (wave & 1) * 64;                 // wave's rows inside block tile
    int c0 = blockIdx.x * 128 + (wave >> 1) * 32;

    v8f acc[4][2];
#pragma unroll
    for (int mi = 0; mi < 4; ++mi)
#pragma unroll
        for (int ni = 0; ni < 2; ++ni) acc[mi][ni] = vzero8f();

    const int nk = K / 32;

    // stage A tile for k-step 0: 128 rows x 32 f16 = 512 chunks of 16B,
    // 256 threads x 2 chunks each
#pragma unroll
    for (int i = 0; i < 2; ++i) {
        int idx  = tid + i * 256;
        int row  = idx >> 2;
        int col8 = (idx & 3) * 8;
        copy16_g2l(A + (size_t)(rowbase + row) * lda + col8,
                   &lbufA[0][row * 32 + col8]);
    }

    for (int kt = 0; kt < nk; ++kt) {
        int cur = kt & 1;
        async_wait_all();
        __syncthreads();

        if (kt + 1 < nk) {
            int k0n = (kt + 1) * 32;
#pragma unroll
            for (int i = 0; i < 2; ++i) {
                int idx  = tid + i * 256;
                int row  = idx >> 2;
                int col8 = (idx & 3) * 8;
                copy16_g2l(A + (size_t)(rowbase + row) * lda + k0n + col8,
                           &lbufA[cur ^ 1][row * 32 + col8]);
            }
            // prefetch next B tile rows into L2
            __builtin_prefetch(Bm + (size_t)(c0 + (lane & 15)) * ldb + k0n, 0, 1);
        }

        int k0 = kt * 32;
        v16h b0 = load_b_frag(Bm, ldb, c0, k0, lane);
        v16h b1 = load_b_frag(Bm, ldb, c0 + 16, k0, lane);
#pragma unroll
        for (int mi = 0; mi < 4; ++mi) {
            v16h a = load_a_frag_lds(&lbufA[cur][0], r0 + 16 * mi, lane);
            acc[mi][0] = wmma16x16x32(a, b0, acc[mi][0]);
            acc[mi][1] = wmma16x16x32(a, b1, acc[mi][1]);
        }
    }

    // C/D layout: element r -> M = r + 8*(lane>=16), N = lane&15
    int n  = lane & 15;
    int mo = (lane >> 4) << 3;
#pragma unroll
    for (int mi = 0; mi < 4; ++mi)
#pragma unroll
        for (int ni = 0; ni < 2; ++ni)
#pragma unroll
            for (int r = 0; r < 8; ++r)
                C[(size_t)(rowbase + r0 + 16 * mi + mo + r) * ldc + c0 +
                  16 * ni + n] = acc[mi][ni][r];
}

// ---------------------------------------------------------------------------
// Kernel 3: per-head RMSNorm (q,k) + NeoX RoPE, f32 qkv -> f16 q/k/v buffers.
// One wave per head-task; 32 tasks per (b,t): q heads 0-15, k 16-23, v 24-31.
// Outputs: q_h [b, h, t, hd], k_h [b, kvh, t, hd], vt_h [b, kvh, hd, t].
// ---------------------------------------------------------------------------
__global__ __launch_bounds__(256) void normrope_kernel(
    const float* __restrict__ qkv, const int* __restrict__ positions,
    const float* __restrict__ rms_w, _Float16* __restrict__ q_h,
    _Float16* __restrict__ k_h, _Float16* __restrict__ vt_h) {
    int lane = threadIdx.x & 31;
    int wave = threadIdx.x >> 5;
    int bt = blockIdx.x;
    int b = bt / TT;
    int t = bt - b * TT;
    float pos = (float)positions[bt];

    for (int it = 0; it < 4; ++it) {
        int task = wave + it * 8;  // 0..31
        const float* src = qkv + (size_t)bt * QKVO + task * HD + lane * 4;
        float4 v4 = *(const float4*)src;
        float vv[4] = {v4.x, v4.y, v4.z, v4.w};

        if (task < 24) {
            // RMSNorm over HD=128 (4 vals/lane * 32 lanes)
            float ss = vv[0] * vv[0] + vv[1] * vv[1] + vv[2] * vv[2] + vv[3] * vv[3];
#pragma unroll
            for (int m = 16; m >= 1; m >>= 1) ss += __shfl_xor(ss, m);
            float inv = rsqrtf(ss * (1.0f / (float)HD) + 1e-6f);

            float nv[4], ov[4];
#pragma unroll
            for (int i = 0; i < 4; ++i)
                nv[i] = vv[i] * inv * rms_w[lane * 4 + i];

            // NeoX RoPE: lanes 0-15 hold x1 (hd 0..63), lanes 16-31 hold x2.
#pragma unroll
            for (int i = 0; i < 4; ++i) {
                float partner = __shfl_xor(nv[i], 16);
                int j = (lane & 15) * 4 + i;  // rotary pair index 0..63
                // inv_freq = 10000^(-j/64) = exp(-j * ln(10000)/64)
                float ang = pos * __expf(-(float)j * 0.14391156f);
                float c = __cosf(ang), s = __sinf(ang);
                ov[i] = (lane < 16) ? (nv[i] * c - partner * s)
                                    : (nv[i] * c + partner * s);
            }
            if (task < 16) {
                _Float16* dst =
                    q_h + ((size_t)(b * NH + task) * TT + t) * HD + lane * 4;
#pragma unroll
                for (int i = 0; i < 4; ++i) dst[i] = (_Float16)ov[i];
            } else {
                int h = task - 16;
                _Float16* dst =
                    k_h + ((size_t)(b * NKV + h) * TT + t) * HD + lane * 4;
#pragma unroll
                for (int i = 0; i < 4; ++i) dst[i] = (_Float16)ov[i];
            }
        } else {
            // V: plain f16 convert, stored transposed [b, kvh, hd, t]
            int h = task - 24;
            _Float16* base = vt_h + ((size_t)(b * NKV + h) * HD) * TT + t;
            int hd = lane * 4;
#pragma unroll
            for (int i = 0; i < 4; ++i)
                base[(size_t)(hd + i) * TT] = (_Float16)vv[i];
        }
    }
}

// ---------------------------------------------------------------------------
// Kernel 4: causal GQA flash attention. One wave per (b, kvh, gi, 16-q tile).
// S = Q*K^T via WMMA (K-dim = HD = 4x32). Online softmax in fp32; P staged
// through wave-private LDS to re-layout into A-fragments; O += P*V via WMMA.
// ---------------------------------------------------------------------------
__global__ __launch_bounds__(256) void attn_fwd_kernel(
    const _Float16* __restrict__ Q, const _Float16* __restrict__ Kk,
    const _Float16* __restrict__ Vt, _Float16* __restrict__ Oa) {
    __shared__ _Float16 lds_p[8][16 * 32];  // 1KB per wave

    int lane = threadIdx.x & 31;
    int wave = threadIdx.x >> 5;
    int task = blockIdx.x * 8 + wave;  // 4096 wave-tasks
    int qtile = task & (TT / 16 - 1);  // 0..127
    int rest = task >> 7;
    int gi  = rest & (GG - 1);
    int kvh = (rest >> 1) & (NKV - 1);
    int b   = rest >> 4;
    int qh  = kvh * GG + gi;
    int q0  = qtile * 16;

    const _Float16* Qb = Q + (size_t)(b * NH + qh) * TT * HD;
    const _Float16* Kb = Kk + (size_t)(b * NKV + kvh) * TT * HD;
    const _Float16* Vb = Vt + (size_t)(b * NKV + kvh) * HD * TT;

    // Preload the Q tile as 4 A-fragments (HD = 4 * 32)
    v16h qf[4];
#pragma unroll
    for (int kk = 0; kk < 4; ++kk)
        qf[kk] = load_a_frag(Qb, HD, q0, kk * 32, lane);

    v8f o[8];
#pragma unroll
    for (int nf = 0; nf < 8; ++nf) o[nf] = vzero8f();
    float mrow[8], lrow[8];
#pragma unroll
    for (int r = 0; r < 8; ++r) { mrow[r] = -1e30f; lrow[r] = 0.f; }

    const float scale = 0.08838834764831845f;  // 1/sqrt(128)
    int n  = lane & 15;
    int mo = (lane >> 4) << 3;

    for (int j0 = 0; j0 <= q0 + 15; j0 += 32) {
        // ---- S = Q * K^T for 32 keys (two 16-key N tiles) ----
        v8f s0 = vzero8f(), s1 = vzero8f();
#pragma unroll
        for (int kk = 0; kk < 4; ++kk) {
            v16h kb0 = load_b_frag(Kb, HD, j0, kk * 32, lane);
            v16h kb1 = load_b_frag(Kb, HD, j0 + 16, kk * 32, lane);
            s0 = wmma16x16x32(qf[kk], kb0, s0);
            s1 = wmma16x16x32(qf[kk], kb1, s1);
        }

        // ---- causal mask + scale, per-row block max ----
        float sv0[8], sv1[8], bmax[8];
#pragma unroll
        for (int r = 0; r < 8; ++r) {
            int m = q0 + mo + r;  // global query index of this row
            sv0[r] = (j0 + n <= m) ? s0[r] * scale : -1e30f;
            sv1[r] = (j0 + 16 + n <= m) ? s1[r] * scale : -1e30f;
            float mx = fmaxf(sv0[r], sv1[r]);
            mx = fmaxf(mx, __shfl_xor(mx, 1));
            mx = fmaxf(mx, __shfl_xor(mx, 2));
            mx = fmaxf(mx, __shfl_xor(mx, 4));
            mx = fmaxf(mx, __shfl_xor(mx, 8));
            bmax[r] = mx;
        }

        // ---- online softmax update; stage P (f16) into LDS ----
        float alpha[8];
#pragma unroll
        for (int r = 0; r < 8; ++r) {
            float mnew = fmaxf(mrow[r], bmax[r]);
            alpha[r] = expf(mrow[r] - mnew);
            mrow[r] = mnew;
            float p0 = expf(sv0[r] - mnew);
            float p1 = expf(sv1[r] - mnew);
            lds_p[wave][(mo + r) * 32 + n]      = (_Float16)p0;
            lds_p[wave][(mo + r) * 32 + 16 + n] = (_Float16)p1;
            float sum = p0 + p1;
            sum += __shfl_xor(sum, 1);
            sum += __shfl_xor(sum, 2);
            sum += __shfl_xor(sum, 4);
            sum += __shfl_xor(sum, 8);
            lrow[r] = lrow[r] * alpha[r] + sum;
        }

        // ---- rescale O accumulators ----
#pragma unroll
        for (int nf = 0; nf < 8; ++nf)
#pragma unroll
            for (int r = 0; r < 8; ++r) o[nf][r] *= alpha[r];

        // ---- O += P * V  (P: 16x32 from LDS; V: transposed, contiguous) ----
        v16h pf = load_a_frag_lds(&lds_p[wave][0], 0, lane);
#pragma unroll
        for (int nf = 0; nf < 8; ++nf) {
            v16h vb = load_b_frag(Vb, TT, nf * 16, j0, lane);
            o[nf] = wmma16x16x32(pf, vb, o[nf]);
        }
    }

    // ---- normalize and store f16 for the O-projection GEMM ----
#pragma unroll
    for (int nf = 0; nf < 8; ++nf) {
#pragma unroll
        for (int r = 0; r < 8; ++r) {
            int t = q0 + mo + r;
            float val = o[nf][r] / lrow[r];
            Oa[((size_t)(b * TT + t)) * QS + qh * HD + nf * 16 + n] =
                (_Float16)val;
        }
    }
}

// ---------------------------------------------------------------------------
// Host launcher
// ---------------------------------------------------------------------------
extern "C" void kernel_launch(void* const* d_in, const int* in_sizes, int n_in,
                              void* d_out, int out_size, void* d_ws,
                              size_t ws_size, hipStream_t stream) {
    const int*   positions = (const int*)d_in[0];
    const float* hidden    = (const float*)d_in[1];
    const float* w_qkv     = (const float*)d_in[2];
    const float* w_o       = (const float*)d_in[3];
    const float* rms_w     = (const float*)d_in[4];
    float*       out       = (float*)d_out;

    char* ws = (char*)d_ws;
    size_t off = 0;
    auto alloc = [&](size_t bytes) -> void* {
        void* p = ws + off;
        off = (off + bytes + 255) & ~(size_t)255;
        return p;
    };

    _Float16* wqkv_h = (_Float16*)alloc((size_t)QKVO * HID * 2);
    _Float16* wo_h   = (_Float16*)alloc((size_t)HID * QS * 2);
    _Float16* x_h    = (_Float16*)alloc((size_t)MROWS * HID * 2);
    float*    qkv_f  = (float*)alloc((size_t)MROWS * QKVO * 4);
    _Float16* q_h    = (_Float16*)alloc((size_t)BB * NH * TT * HD * 2);
    _Float16* k_h    = (_Float16*)alloc((size_t)BB * NKV * TT * HD * 2);
    _Float16* vt_h   = (_Float16*)alloc((size_t)BB * NKV * HD * TT * 2);
    _Float16* attn_h = (_Float16*)alloc((size_t)MROWS * QS * 2);

    // 1) precision conversion
    f32_to_f16_kernel<<<4096, 256, 0, stream>>>(w_qkv, wqkv_h, QKVO * HID);
    f32_to_f16_kernel<<<4096, 256, 0, stream>>>(w_o, wo_h, HID * QS);
    f32_to_f16_kernel<<<4096, 256, 0, stream>>>(hidden, x_h, MROWS * HID);

    // 2) QKV projection: [4096,2048] x [4096,2048]^T -> [4096,4096] f32
    gemm_nt_f16_kernel<<<dim3(QKVO / 128, MROWS / 128), 256, 0, stream>>>(
        x_h, wqkv_h, qkv_f, HID, HID, HID, QKVO);

    // 3) QK RMSNorm + RoPE, split/convert to attention layouts
    normrope_kernel<<<MROWS, 256, 0, stream>>>(qkv_f, positions, rms_w, q_h,
                                               k_h, vt_h);

    // 4) causal GQA flash attention (4096 wave-tasks / 8 waves per block)
    attn_fwd_kernel<<<(BB * NKV * GG * (TT / 16)) / 8, 256, 0, stream>>>(
        q_h, k_h, vt_h, attn_h);

    // 5) output projection: [4096,2048] x [2048,2048]^T -> d_out f32
    gemm_nt_f16_kernel<<<dim3(HID / 128, MROWS / 128), 256, 0, stream>>>(
        attn_h, wo_h, out, QS, QS, QS, HID);
}
